// MaskedAutoregressiveFlow_24532853194797
// MI455X (gfx1250) — matrix operand order repaired and checked by reading, not compiled, not run on hardware
//
#include <hip/hip_runtime.h>

typedef __attribute__((ext_vector_type(16))) _Float16 v16h;
typedef __attribute__((ext_vector_type(8)))  _Float16 v8h;
typedef __attribute__((ext_vector_type(8)))  float    v8f;
typedef __attribute__((ext_vector_type(4)))  float    v4f;

static constexpr int Bsz    = 16384;
static constexpr int Fdim   = 64;
static constexpr int CTXD   = 128;
static constexpr int HID    = 512;
static constexpr int Lnum   = 8;
static constexpr int IN_DIM = 160;   // 32 id features + 128 context

#if __has_builtin(__builtin_amdgcn_global_load_async_to_lds_b128)
#define MAF_ASYNC_LDS 1
#else
#define MAF_ASYNC_LDS 0
#endif

// Builtin signature (from hipcc diagnostic): param0 = int4 addrspace(1)*
// (prints as "int4 __device__*"), payload type = gcc-vector int4.
typedef int maf_v4i __attribute__((vector_size(16)));
#define MAF_GP(p) ((__attribute__((address_space(1))) maf_v4i*)(unsigned long long)(p))
#define MAF_LP(p) ((__attribute__((address_space(3))) maf_v4i*)(unsigned int)(unsigned long long)(p))

static __device__ __forceinline__ void maf_wait_async0() {
#if __has_builtin(__builtin_amdgcn_s_wait_asynccnt)
    __builtin_amdgcn_s_wait_asynccnt(0);
#else
    asm volatile("s_wait_asynccnt 0x0" ::: "memory");
#endif
}

// ---------------------------------------------------------------------------
// prep: x = clip(inputs, -1, 1); logdet = 0
// ---------------------------------------------------------------------------
__global__ __launch_bounds__(256) void maf_prep_kernel(
    const float* __restrict__ in, float* __restrict__ x,
    float* __restrict__ logdet, int total)
{
    int i = blockIdx.x * 256 + threadIdx.x;
    if (i < total) x[i] = fminf(1.f, fmaxf(-1.f, in[i]));
    if (i < Bsz)   logdet[i] = 0.f;
}

// ---------------------------------------------------------------------------
// one-time per call: convert fp32 weights [count][K][N] -> fp16 transposed
// [count][N][K] so GEMM B-tiles are contiguous along K.
// ---------------------------------------------------------------------------
__global__ __launch_bounds__(256) void maf_wconv_kernel(
    const float* __restrict__ W, _Float16* __restrict__ Wt,
    int K, int N, int total)
{
    int i = blockIdx.x * 256 + threadIdx.x;
    if (i >= total) return;
    int kn = K * N;
    int l  = i / kn;
    int r  = i % kn;
    int k  = r / N;
    int n  = r % N;
    Wt[(size_t)l * kn + (size_t)n * K + k] = (_Float16)W[i];
}

// ---------------------------------------------------------------------------
// build GEMM input A0 (B x 160, fp16) = [ x[:, id_idx] | context ]
// ---------------------------------------------------------------------------
__global__ __launch_bounds__(256) void maf_build_in_kernel(
    const float* __restrict__ x, const float* __restrict__ ctx,
    _Float16* __restrict__ A0, int idoff)
{
    int i   = blockIdx.x * 256 + threadIdx.x;   // grid covers B*IN_DIM exactly
    int row = i / IN_DIM;
    int col = i % IN_DIM;
    float v = (col < 32) ? x[row * Fdim + 2 * col + idoff]
                         : ctx[row * CTXD + (col - 32)];
    A0[i] = (_Float16)v;
}

// ---------------------------------------------------------------------------
// WMMA GEMM:  Out = act(A_f16[MxK] @ Wt_f16[NxK]^T + bias)
//  - block tile BM x BN, 8 waves, wave tile WM x WN, BK = 32
//  - double-buffered LDS staging; async global->LDS copies when available
//  - epilogue staged through LDS -> coalesced b128 global stores
// ---------------------------------------------------------------------------
template<int BM, int BN, int WM, int WN, bool RELU, bool OUT_F16>
__global__ __launch_bounds__(256) void maf_gemm_kernel(
    const _Float16* __restrict__ A,    // M x K row-major fp16
    const _Float16* __restrict__ Wt,   // N x K row-major fp16 (pre-transposed)
    const float*    __restrict__ bias, // N
    void*           __restrict__ Out,  // M x N (fp16 or fp32)
    int M, int K, int N)
{
    static_assert(BM == 128, "A-tile loader assumes BM==128");
    constexpr int WAVES_M = BM / WM;
    constexpr int WAVES_N = BN / WN;
    static_assert(WAVES_M * WAVES_N == 8, "8 waves per block");
    constexpr int TM  = WM / 16;
    constexpr int TN  = WN / 16;
    constexpr int LDA = 40;                 // halfs per LDS row (32 + 8 pad)
    constexpr int LDB = 40;
    constexpr int BUF_H = BM * LDA + BN * LDB;     // halfs per stage buffer

    // B-tile staging: BN*32 halfs over 256 threads
    constexpr int B_HALFS  = BN * 32 / 256;        // 16 (BN=128) / 8 (BN=64)
    constexpr int B_VEC    = B_HALFS / 8;          // v8h chunks per thread
    constexpr int TPB_ROW  = 32 / B_HALFS;         // threads per B row

    constexpr size_t STAGE_BYTES = (size_t)2 * 2 * BUF_H;
    constexpr size_t C_BYTES = OUT_F16 ? (size_t)BM * (BN + 8) * 2
                                       : (size_t)BM * (BN + 4) * 4;
    constexpr size_t SMEM_BYTES = STAGE_BYTES > C_BYTES ? STAGE_BYTES : C_BYTES;
    __shared__ __align__(16) char smem[SMEM_BYTES];
    _Float16* stage = (_Float16*)smem;

    const int tid  = threadIdx.x;
    const int wave = tid >> 5;
    const int lane = tid & 31;
    const int wm   = wave / WAVES_N;
    const int wn   = wave % WAVES_N;
    const int m15  = lane & 15;             // A: row-in-tile / B,C: col-in-tile
    const int hk   = lane >> 4;             // K-half selector

    const int block_m = blockIdx.y * BM;
    const int block_n = blockIdx.x * BN;

    // staging thread mapping
    const int rA = tid >> 1;                 // A row (0..127)
    const int cA = (tid & 1) * 16;           // A col (0 / 16)
    const int nB = tid / TPB_ROW;            // B row (n)
    const int cB = (tid % TPB_ROW) * B_HALFS;

    const _Float16* srcA_base = A  + (size_t)(block_m + rA) * K + cA;
    const _Float16* srcB_base = Wt + (size_t)(block_n + nB) * K + cB;

    v8f acc[TM][TN] = {};
    union V16 { v16h v; v8h h[2]; };

#if MAF_ASYNC_LDS
    auto issue_stage = [&](int k0, int buf) {
        _Float16* As = stage + buf * BUF_H;
        _Float16* Bs = As + BM * LDA;
        const _Float16* sa = srcA_base + k0;
        __builtin_amdgcn_global_load_async_to_lds_b128(
            MAF_GP(sa),     MAF_LP(&As[rA * LDA + cA]),     0, 0);
        __builtin_amdgcn_global_load_async_to_lds_b128(
            MAF_GP(sa + 8), MAF_LP(&As[rA * LDA + cA + 8]), 0, 0);
        const _Float16* sb = srcB_base + k0;
        #pragma unroll
        for (int j = 0; j < B_VEC; ++j)
            __builtin_amdgcn_global_load_async_to_lds_b128(
                MAF_GP(sb + 8 * j), MAF_LP(&Bs[nB * LDB + cB + 8 * j]), 0, 0);
    };
#else
    v8h ar0, ar1, br[2];
    auto load_global = [&](int k0) {
        const _Float16* sa = srcA_base + k0;
        ar0 = *(const v8h*)(sa);
        ar1 = *(const v8h*)(sa + 8);
        const _Float16* sb = srcB_base + k0;
        #pragma unroll
        for (int j = 0; j < B_VEC; ++j) br[j] = *(const v8h*)(sb + 8 * j);
    };
    auto store_stage = [&](int buf) {
        _Float16* As = stage + buf * BUF_H;
        _Float16* Bs = As + BM * LDA;
        *(v8h*)(&As[rA * LDA + cA])     = ar0;
        *(v8h*)(&As[rA * LDA + cA + 8]) = ar1;
        #pragma unroll
        for (int j = 0; j < B_VEC; ++j) *(v8h*)(&Bs[nB * LDB + cB + 8 * j]) = br[j];
    };
#endif

    const int nsteps = K >> 5;
#if MAF_ASYNC_LDS
    issue_stage(0, 0);
    maf_wait_async0();
    __syncthreads();
#else
    load_global(0);
    store_stage(0);
    __syncthreads();
#endif

    for (int s = 0; s < nsteps; ++s) {
        const int cur = s & 1;
#if MAF_ASYNC_LDS
        if (s + 1 < nsteps) issue_stage((s + 1) << 5, cur ^ 1);
#else
        if (s + 1 < nsteps) load_global((s + 1) << 5);
#endif

        const _Float16* As = stage + cur * BUF_H;
        const _Float16* Bs = As + BM * LDA;

        V16 af[TM], bf[TN];
        #pragma unroll
        for (int tm = 0; tm < TM; ++tm) {
            // A layout: lane 16*hk+m; elems 0..7 -> K=8hk+e, 8..15 -> 16+8hk+e'
            const _Float16* arw = As + (wm * WM + tm * 16 + m15) * LDA;
            af[tm].h[0] = *(const v8h*)(arw + 8 * hk);
            af[tm].h[1] = *(const v8h*)(arw + 16 + 8 * hk);
        }
        #pragma unroll
        for (int tn = 0; tn < TN; ++tn) {
            // B layout: lane 16*hk+n; elems 0..7 -> K=8hk+e, 8..15 -> 16+8hk+e'
            const _Float16* brw = Bs + (wn * WN + tn * 16 + m15) * LDB;
            bf[tn].h[0] = *(const v8h*)(brw + 8 * hk);
            bf[tn].h[1] = *(const v8h*)(brw + 16 + 8 * hk);
        }
        #pragma unroll
        for (int tm = 0; tm < TM; ++tm)
            #pragma unroll
            for (int tn = 0; tn < TN; ++tn)
                acc[tm][tn] = __builtin_amdgcn_wmma_f32_16x16x32_f16(
                    false, af[tm].v, false, bf[tn].v, (short)0, acc[tm][tn],
                    false, false);

#if MAF_ASYNC_LDS
        if (s + 1 < nsteps) maf_wait_async0();
#else
        if (s + 1 < nsteps) store_stage(cur ^ 1);
#endif
        __syncthreads();
    }

    // ---- epilogue: bias + act into LDS C tile, then coalesced b128 stores --
    if constexpr (OUT_F16) {
        constexpr int LDC = BN + 8;
        _Float16* Cs = (_Float16*)smem;
        #pragma unroll
        for (int tm = 0; tm < TM; ++tm) {
            #pragma unroll
            for (int tn = 0; tn < TN; ++tn) {
                int rl0 = wm * WM + tm * 16 + 8 * hk;  // D: M = r + 8*half
                int cl  = wn * WN + tn * 16 + m15;     // D: N = lane&15
                float bv = bias[block_n + cl];
                #pragma unroll
                for (int r = 0; r < 8; ++r) {
                    float v = acc[tm][tn][r] + bv;
                    if (RELU) v = fmaxf(v, 0.f);
                    Cs[(rl0 + r) * LDC + cl] = (_Float16)v;
                }
            }
        }
        __syncthreads();
        constexpr int CH = BM * BN / 2048;             // v8h chunks per thread
        _Float16* outp = (_Float16*)Out;
        #pragma unroll
        for (int e = 0; e < CH; ++e) {
            int flat8 = tid + e * 256;
            int row   = flat8 / (BN / 8);
            int col0  = (flat8 % (BN / 8)) * 8;
            v8h v = *(const v8h*)(Cs + row * LDC + col0);
            *(v8h*)(outp + (size_t)(block_m + row) * N + block_n + col0) = v;
        }
    } else {
        constexpr int LDCf = BN + 4;
        float* Cs = (float*)smem;
        #pragma unroll
        for (int tm = 0; tm < TM; ++tm) {
            #pragma unroll
            for (int tn = 0; tn < TN; ++tn) {
                int rl0 = wm * WM + tm * 16 + 8 * hk;
                int cl  = wn * WN + tn * 16 + m15;
                float bv = bias[block_n + cl];
                #pragma unroll
                for (int r = 0; r < 8; ++r) {
                    float v = acc[tm][tn][r] + bv;
                    if (RELU) v = fmaxf(v, 0.f);
                    Cs[(rl0 + r) * LDCf + cl] = v;
                }
            }
        }
        __syncthreads();
        constexpr int CH = BM * BN / 1024;             // v4f chunks per thread
        float* outp = (float*)Out;
        #pragma unroll
        for (int e = 0; e < CH; ++e) {
            int flat4 = tid + e * 256;
            int row   = flat4 / (BN / 4);
            int col0  = (flat4 % (BN / 4)) * 4;
            v4f v = *(const v4f*)(Cs + row * LDCf + col0);
            *(v4f*)(outp + (size_t)(block_m + row) * N + block_n + col0) = v;
        }
    }
}

// ---------------------------------------------------------------------------
// coupling: scale/shift transform, logdet accumulation, permutation gather
// ---------------------------------------------------------------------------
__global__ __launch_bounds__(256) void maf_coupling_kernel(
    const float* __restrict__ p, const float* __restrict__ xin,
    float* __restrict__ xout, float* __restrict__ logdet,
    const int* __restrict__ perm, int toff)
{
    int row = blockIdx.x * 256 + threadIdx.x;
    if (row >= Bsz) return;
    const float* pr = p   + (size_t)row * Fdim;
    const float* xr = xin + (size_t)row * Fdim;
    float upd[Fdim];
    int ioff = 1 - toff;
    float ld = 0.f;
    #pragma unroll
    for (int c = 0; c < 32; ++c) {
        float shift = pr[c];
        float s = 1.f / (1.f + expf(-(pr[32 + c] + 2.f))) + 0.001f;
        ld += logf(s);
        upd[2 * c + toff] = xr[2 * c + toff] * s + shift;
        upd[2 * c + ioff] = xr[2 * c + ioff];
    }
    logdet[row] += ld;
    float* xo = xout + (size_t)row * Fdim;
    if (perm) {
        for (int f = 0; f < Fdim; ++f) xo[f] = upd[perm[f]];
    } else {
        for (int f = 0; f < Fdim; ++f) xo[f] = upd[f];
    }
}

// ---------------------------------------------------------------------------
// final: out = [clip(x,-1,1) (B*F) | logdet (B)]
// ---------------------------------------------------------------------------
__global__ __launch_bounds__(256) void maf_final_kernel(
    const float* __restrict__ x, const float* __restrict__ logdet,
    float* __restrict__ out, int total)
{
    int i = blockIdx.x * 256 + threadIdx.x;
    if (i < total) out[i] = fminf(1.f, fmaxf(-1.f, x[i]));
    if (i < Bsz)   out[total + i] = logdet[i];
}

// ---------------------------------------------------------------------------
extern "C" void kernel_launch(void* const* d_in, const int* in_sizes, int n_in,
                              void* d_out, int out_size, void* d_ws, size_t ws_size,
                              hipStream_t stream) {
    (void)in_sizes; (void)n_in; (void)out_size; (void)ws_size;
    const float* inputs  = (const float*)d_in[0];
    const float* context = (const float*)d_in[1];
    const float* W_in    = (const float*)d_in[2];
    const float* b_in    = (const float*)d_in[3];
    const float* W_h     = (const float*)d_in[4];
    const float* b_h     = (const float*)d_in[5];
    const float* W_out   = (const float*)d_in[6];
    const float* b_out   = (const float*)d_in[7];
    const int*   perms   = (const int*)d_in[8];

    char* ws = (char*)d_ws;
    float*    xA     = (float*)(ws + 0);                    // B*64 f32
    float*    xB     = (float*)(ws + 4194304);              // B*64 f32
    float*    logdet = (float*)(ws + 8388608);              // B f32
    _Float16* A0     = (_Float16*)(ws + 8454144);           // B*160 f16
    _Float16* hA     = (_Float16*)(ws + 13697024);          // B*512 f16
    _Float16* hB     = (_Float16*)(ws + 30474240);          // B*512 f16
    float*    p      = (float*)(ws + 47251456);             // B*64 f32
    _Float16* Wt_in  = (_Float16*)(ws + 51445760);          // L*512*160 f16
    _Float16* Wt_h   = (_Float16*)(ws + 52756480);          // L*2*512*512 f16
    _Float16* Wt_out = (_Float16*)(ws + 61145088);          // L*64*512 f16

    // one-time weight convert + transpose (fp32 KxN -> fp16 NxK)
    maf_wconv_kernel<<<(Lnum * IN_DIM * HID) / 256, 256, 0, stream>>>(
        W_in, Wt_in, IN_DIM, HID, Lnum * IN_DIM * HID);
    maf_wconv_kernel<<<(Lnum * 2 * HID * HID) / 256, 256, 0, stream>>>(
        W_h, Wt_h, HID, HID, Lnum * 2 * HID * HID);
    maf_wconv_kernel<<<(Lnum * HID * Fdim) / 256, 256, 0, stream>>>(
        W_out, Wt_out, HID, Fdim, Lnum * HID * Fdim);

    maf_prep_kernel<<<(Bsz * Fdim) / 256, 256, 0, stream>>>(
        inputs, xA, logdet, Bsz * Fdim);

    float* xcur = xA;
    float* xnxt = xB;
    const dim3 gHid(HID / 128, Bsz / 128);   // (4, 128)
    const dim3 gOut(1, Bsz / 128);           // (1, 128)

    for (int i = 0; i < Lnum; ++i) {
        const int idoff = (i % 2 == 0) ? 1 : 0;  // id indices (odd on even layers)
        const int toff  = (i % 2 == 0) ? 0 : 1;  // transformed indices

        maf_build_in_kernel<<<(Bsz * IN_DIM) / 256, 256, 0, stream>>>(
            xcur, context, A0, idoff);

        // h = relu(A0 @ W_in[i] + b_in[i])                 K=160, N=512
        maf_gemm_kernel<128, 128, 32, 64, true, true><<<gHid, 256, 0, stream>>>(
            A0, Wt_in + (size_t)i * IN_DIM * HID, b_in + i * HID, hA,
            Bsz, IN_DIM, HID);
        // h = relu(h @ W_h[i,0] + b_h[i,0])                K=512, N=512
        maf_gemm_kernel<128, 128, 32, 64, true, true><<<gHid, 256, 0, stream>>>(
            hA, Wt_h + (size_t)(i * 2 + 0) * HID * HID, b_h + (i * 2 + 0) * HID,
            hB, Bsz, HID, HID);
        // h = relu(h @ W_h[i,1] + b_h[i,1])                K=512, N=512
        maf_gemm_kernel<128, 128, 32, 64, true, true><<<gHid, 256, 0, stream>>>(
            hB, Wt_h + (size_t)(i * 2 + 1) * HID * HID, b_h + (i * 2 + 1) * HID,
            hA, Bsz, HID, HID);
        // p = h @ W_out[i] + b_out[i]                      K=512, N=64
        maf_gemm_kernel<128, 64, 16, 64, false, false><<<gOut, 256, 0, stream>>>(
            hA, Wt_out + (size_t)i * HID * Fdim, b_out + i * Fdim, p,
            Bsz, HID, Fdim);

        const int* perm = (i < Lnum - 1) ? (perms + i * Fdim) : nullptr;
        maf_coupling_kernel<<<Bsz / 256, 256, 0, stream>>>(
            p, xcur, xnxt, logdet, perm, toff);

        float* t = xcur; xcur = xnxt; xnxt = t;
    }

    maf_final_kernel<<<(Bsz * Fdim) / 256, 256, 0, stream>>>(
        xcur, logdet, (float*)d_out, Bsz * Fdim);
}
